// SparseMoEFFN_20813411516573
// MI455X (gfx1250) — matrix operand order repaired and checked by reading, not compiled, not run on hardware
//
#include <hip/hip_runtime.h>
#include <hip/hip_bf16.h>
#include <math.h>

// ---------------- problem constants ----------------
#define TOK     16384          // B*K tokens
#define DM      1024           // d_model
#define FF      4096           // d_ff
#define NE      8              // experts
#define SLOTS   (TOK * 2)      // top-2 token-slots
#define TILE    128            // M/N tile
#define KC      32             // K chunk per WMMA (bf16)
#define PADROWS 33792          // SLOTS + NE*(TILE-1) rounded to TILE multiple

typedef __attribute__((ext_vector_type(16))) __bf16 v16bf;
typedef __attribute__((ext_vector_type(8)))  float  v8f;

union FragU { v16bf v; uint4 q[2]; };

// ---------------- utility kernels ----------------
__global__ void k_zero_f32(float* __restrict__ p, int n) {
    int i = blockIdx.x * blockDim.x + threadIdx.x;
    if (i < n) p[i] = 0.0f;
}

__global__ void k_init_ctrl(int* __restrict__ row_tok, int* __restrict__ counts,
                            float* __restrict__ psum, int* __restrict__ cursors) {
    int i = blockIdx.x * blockDim.x + threadIdx.x;
    if (i < PADROWS) row_tok[i] = -1;
    if (i < NE) { counts[i] = 0; psum[i] = 0.0f; cursors[i] = 0; }
}

__global__ void k_cast_bf16(const float* __restrict__ s, __bf16* __restrict__ d, int n4) {
    int i = blockIdx.x * blockDim.x + threadIdx.x;
    if (i >= n4) return;
    float4 f = ((const float4*)s)[i];
    union { __bf16 b[4]; uint2 u; } r;
    r.b[0] = (__bf16)f.x; r.b[1] = (__bf16)f.y;
    r.b[2] = (__bf16)f.z; r.b[3] = (__bf16)f.w;
    ((uint2*)d)[i] = r.u;
}

// ---------------- router: softmax + top-2 ----------------
__global__ __launch_bounds__(256)
void k_router(const float* __restrict__ x, const float* __restrict__ wr,
              int* __restrict__ topki, float* __restrict__ topkp,
              int* __restrict__ counts, float* __restrict__ psum) {
    __shared__ float sP[NE];
    __shared__ int   sC[NE];
    int tid = threadIdx.x;
    if (tid < NE) { sP[tid] = 0.0f; sC[tid] = 0; }
    __syncthreads();

    int wv = tid >> 5, lane = tid & 31;
    int t = blockIdx.x * 8 + wv;        // wave per token

    float part[NE];
#pragma unroll
    for (int e = 0; e < NE; ++e) part[e] = 0.0f;

    const float* xr = x + (size_t)t * DM;
    for (int i = 0; i < DM / 32; ++i) {
        float xv = xr[i * 32 + lane];
        const float* w = wr + (size_t)(i * 32 + lane) * NE;
#pragma unroll
        for (int e = 0; e < NE; ++e) part[e] += xv * w[e];
    }
#pragma unroll
    for (int e = 0; e < NE; ++e) {
#pragma unroll
        for (int off = 16; off >= 1; off >>= 1)
            part[e] += __shfl_xor(part[e], off, 32);
    }

    if (lane == 0) {
        float m = part[0];
#pragma unroll
        for (int e = 1; e < NE; ++e) m = fmaxf(m, part[e]);
        float p[NE], s = 0.0f;
#pragma unroll
        for (int e = 0; e < NE; ++e) { p[e] = expf(part[e] - m); s += p[e]; }
        float inv = 1.0f / s;
#pragma unroll
        for (int e = 0; e < NE; ++e) p[e] *= inv;

        int i1 = 0; float v1 = p[0];
#pragma unroll
        for (int e = 1; e < NE; ++e) if (p[e] > v1) { v1 = p[e]; i1 = e; }
        int i2 = -1; float v2 = -1.0f;
#pragma unroll
        for (int e = 0; e < NE; ++e) if (e != i1 && p[e] > v2) { v2 = p[e]; i2 = e; }

        float pn = 1.0f / (v1 + v2);
        topki[2 * t]     = i1;  topki[2 * t + 1] = i2;
        topkp[2 * t]     = v1 * pn;
        topkp[2 * t + 1] = v2 * pn;

        atomicAdd(&sC[i1], 1);
        atomicAdd(&sC[i2], 1);
#pragma unroll
        for (int e = 0; e < NE; ++e) atomicAdd(&sP[e], p[e]);
    }
    __syncthreads();
    if (tid < NE) {
        atomicAdd(&counts[tid], sC[tid]);
        atomicAdd(&psum[tid], sP[tid]);
    }
}

// ---------------- scan: padded offsets + aux loss ----------------
__global__ void k_scan(const int* __restrict__ counts, const float* __restrict__ psum,
                       int* __restrict__ offpad, float* __restrict__ aux_out) {
    if (threadIdx.x == 0) {
        int acc = 0; offpad[0] = 0;
        float aux = 0.0f;
        for (int e = 0; e < NE; ++e) {
            int cap = ((counts[e] + TILE - 1) / TILE) * TILE;
            acc += cap; offpad[e + 1] = acc;
            float f = (float)counts[e] / (float)SLOTS;
            float P = psum[e] / (float)TOK;
            aux += f * P;
        }
        *aux_out = (float)NE * 0.01f * aux;
    }
}

// ---------------- scatter token-slots into expert row lists ----------------
__global__ void k_scatter(const int* __restrict__ topki, const float* __restrict__ topkp,
                          const int* __restrict__ offpad, int* __restrict__ cursors,
                          int* __restrict__ rowtok, float* __restrict__ roww) {
    int t = blockIdx.x * blockDim.x + threadIdx.x;
    if (t >= TOK) return;
#pragma unroll
    for (int s = 0; s < 2; ++s) {
        int e = topki[2 * t + s];
        int pos = atomicAdd(&cursors[e], 1);
        int row = offpad[e] + pos;
        rowtok[row] = t;
        roww[row]   = topkp[2 * t + s];
    }
}

// ---------------- GEMM1: H = gelu(Xg @ W1[e] + b1[e]) ----------------
__global__ __launch_bounds__(256)
void k_ffn1(const __bf16* __restrict__ xb, const __bf16* __restrict__ w1b,
            const float* __restrict__ b1, const int* __restrict__ offpad,
            const int* __restrict__ rowtok, __bf16* __restrict__ hb) {
    __shared__ __bf16 As[2][TILE * KC];   // 128x32 gathered X, row-major (stride KC)
    __shared__ __bf16 Bs[2][KC * TILE];   // 32x128 W1 tile, row-major (stride TILE)
    __shared__ int sE;

    int tid = threadIdx.x;
    int rowBase = blockIdx.x * TILE;
    if (tid == 0) {
        int e = -1;
        if (rowBase < offpad[NE])
            for (int i = 0; i < NE; ++i)
                if (rowBase >= offpad[i] && rowBase < offpad[i + 1]) { e = i; break; }
        sE = e;
    }
    __syncthreads();
    int e = sE;
    if (e < 0) return;

    int colBase = blockIdx.y * TILE;
    int lane = tid & 31, wv = tid >> 5;
    int wm = wv >> 1, wn = wv & 1;       // 4x2 wave grid: 32 rows x 64 cols per wave

    // global-load assignments (16 bf16 = 32B per thread per stage)
    int ar = tid >> 1, ah = tid & 1;
    int tok = rowtok[rowBase + ar];
    const __bf16* aSrc = xb + (size_t)(tok < 0 ? 0 : tok) * DM + ah * 16;
    int bk = tid >> 3, bc = (tid & 7) * 16;
    const __bf16* bSrc = w1b + (size_t)e * DM * FF + (size_t)bk * FF + colBase + bc;

    v8f acc[2][4];
    v8f zero = {0, 0, 0, 0, 0, 0, 0, 0};
#pragma unroll
    for (int i = 0; i < 2; ++i)
#pragma unroll
        for (int j = 0; j < 4; ++j) acc[i][j] = zero;

    uint4 aReg = make_uint4(0, 0, 0, 0), bReg;
    if (tok >= 0) aReg = *(const uint4*)aSrc;
    bReg = *(const uint4*)bSrc;
    *(uint4*)&As[0][ar * KC + ah * 16] = aReg;
    *(uint4*)&Bs[0][bk * TILE + bc]    = bReg;
    __syncthreads();

    const int KT = DM / KC;              // 32 K-steps
    for (int kt = 0; kt < KT; ++kt) {
        int cur = kt & 1;
        if (kt + 1 < KT) {
            int k0 = (kt + 1) * KC;
            aReg = make_uint4(0, 0, 0, 0);
            if (tok >= 0) aReg = *(const uint4*)(aSrc + k0);
            bReg = *(const uint4*)(bSrc + (size_t)k0 * FF);
            if (kt + 2 < KT)
                __builtin_prefetch(bSrc + (size_t)(kt + 2) * KC * FF, 0, 1);
        }

        FragU afr[2], bfr[4];
        int kh = (lane >> 4) * 8;        // ISA 16-bit A layout: lane halves own K{0-7,16-23}/{8-15,24-31}
#pragma unroll
        for (int ti = 0; ti < 2; ++ti) {
            int r = wm * 32 + ti * 16 + (lane & 15);
            afr[ti].q[0] = *(const uint4*)&As[cur][r * KC + kh];
            afr[ti].q[1] = *(const uint4*)&As[cur][r * KC + 16 + kh];
        }
#pragma unroll
        for (int tj = 0; tj < 4; ++tj) { // B: lane = N column, K in same table order as A
            int n0 = wn * 64 + tj * 16;
#pragma unroll
            for (int s = 0; s < 16; ++s) {
                int k = (s < 8) ? (kh + s) : (16 + kh + s - 8);
                bfr[tj].v[s] = Bs[cur][k * TILE + n0 + (lane & 15)];
            }
        }
#pragma unroll
        for (int ti = 0; ti < 2; ++ti)
#pragma unroll
            for (int tj = 0; tj < 4; ++tj)
                acc[ti][tj] = __builtin_amdgcn_wmma_f32_16x16x32_bf16(
                    false, afr[ti].v, false, bfr[tj].v,
                    (short)0, acc[ti][tj], false, false);

        __syncthreads();
        if (kt + 1 < KT) {
            int nxt = cur ^ 1;
            *(uint4*)&As[nxt][ar * KC + ah * 16] = aReg;
            *(uint4*)&Bs[nxt][bk * TILE + bc]    = bReg;
            __syncthreads();
        }
    }

    // epilogue: bias + exact GELU -> H (bf16)
#pragma unroll
    for (int ti = 0; ti < 2; ++ti)
#pragma unroll
        for (int tj = 0; tj < 4; ++tj) {
            int m0 = wm * 32 + ti * 16, n0 = wn * 64 + tj * 16;
            int col = colBase + n0 + (lane & 15);
            float bias = b1[e * FF + col];
#pragma unroll
            for (int r = 0; r < 8; ++r) {
                int m = m0 + r + (lane >> 4) * 8;
                size_t row = (size_t)rowBase + m;
                float v = acc[ti][tj][r] + bias;
                float g = 0.5f * v * (1.0f + erff(v * 0.70710678118654752f));
                hb[row * FF + col] = (__bf16)g;
            }
        }
}

// ---------------- GEMM2: out += w * (H @ W2[e] + b2[e]) ----------------
__global__ __launch_bounds__(256)
void k_ffn2(const __bf16* __restrict__ hb, const __bf16* __restrict__ w2b,
            const float* __restrict__ b2, const int* __restrict__ offpad,
            const int* __restrict__ rowtok, const float* __restrict__ roww,
            float* __restrict__ out) {
    __shared__ __bf16 As[2][TILE * KC];
    __shared__ __bf16 Bs[2][KC * TILE];
    __shared__ int sE;

    int tid = threadIdx.x;
    int rowBase = blockIdx.x * TILE;
    if (tid == 0) {
        int e = -1;
        if (rowBase < offpad[NE])
            for (int i = 0; i < NE; ++i)
                if (rowBase >= offpad[i] && rowBase < offpad[i + 1]) { e = i; break; }
        sE = e;
    }
    __syncthreads();
    int e = sE;
    if (e < 0) return;

    int colBase = blockIdx.y * TILE;
    int lane = tid & 31, wv = tid >> 5;
    int wm = wv >> 1, wn = wv & 1;

    int ar = tid >> 1, ah = tid & 1;
    const __bf16* aSrc = hb + (size_t)(rowBase + ar) * FF + ah * 16;
    int bk = tid >> 3, bc = (tid & 7) * 16;
    const __bf16* bSrc = w2b + (size_t)e * FF * DM + (size_t)bk * DM + colBase + bc;

    v8f acc[2][4];
    v8f zero = {0, 0, 0, 0, 0, 0, 0, 0};
#pragma unroll
    for (int i = 0; i < 2; ++i)
#pragma unroll
        for (int j = 0; j < 4; ++j) acc[i][j] = zero;

    uint4 aReg = *(const uint4*)aSrc;
    uint4 bReg = *(const uint4*)bSrc;
    *(uint4*)&As[0][ar * KC + ah * 16] = aReg;
    *(uint4*)&Bs[0][bk * TILE + bc]    = bReg;
    __syncthreads();

    const int KT = FF / KC;              // 128 K-steps
    for (int kt = 0; kt < KT; ++kt) {
        int cur = kt & 1;
        if (kt + 1 < KT) {
            int k0 = (kt + 1) * KC;
            aReg = *(const uint4*)(aSrc + k0);
            bReg = *(const uint4*)(bSrc + (size_t)k0 * DM);
            if (kt + 2 < KT)
                __builtin_prefetch(bSrc + (size_t)(kt + 2) * KC * DM, 0, 1);
        }

        FragU afr[2], bfr[4];
        int kh = (lane >> 4) * 8;
#pragma unroll
        for (int ti = 0; ti < 2; ++ti) {
            int r = wm * 32 + ti * 16 + (lane & 15);
            afr[ti].q[0] = *(const uint4*)&As[cur][r * KC + kh];
            afr[ti].q[1] = *(const uint4*)&As[cur][r * KC + 16 + kh];
        }
#pragma unroll
        for (int tj = 0; tj < 4; ++tj) {
            int n0 = wn * 64 + tj * 16;
#pragma unroll
            for (int s = 0; s < 16; ++s) {
                int k = (s < 8) ? (kh + s) : (16 + kh + s - 8);
                bfr[tj].v[s] = Bs[cur][k * TILE + n0 + (lane & 15)];
            }
        }
#pragma unroll
        for (int ti = 0; ti < 2; ++ti)
#pragma unroll
            for (int tj = 0; tj < 4; ++tj)
                acc[ti][tj] = __builtin_amdgcn_wmma_f32_16x16x32_bf16(
                    false, afr[ti].v, false, bfr[tj].v,
                    (short)0, acc[ti][tj], false, false);

        __syncthreads();
        if (kt + 1 < KT) {
            int nxt = cur ^ 1;
            *(uint4*)&As[nxt][ar * KC + ah * 16] = aReg;
            *(uint4*)&Bs[nxt][bk * TILE + bc]    = bReg;
            __syncthreads();
        }
    }

    // epilogue: bias + combine-weight scaled scatter-add (token -1 = padding)
#pragma unroll
    for (int ti = 0; ti < 2; ++ti)
#pragma unroll
        for (int tj = 0; tj < 4; ++tj) {
            int m0 = wm * 32 + ti * 16, n0 = wn * 64 + tj * 16;
            int col = colBase + n0 + (lane & 15);
            float bias = b2[e * DM + col];
#pragma unroll
            for (int r = 0; r < 8; ++r) {
                int m = m0 + r + (lane >> 4) * 8;
                int row = rowBase + m;
                int tokid = rowtok[row];
                if (tokid >= 0) {
                    float wgt = roww[row];
                    float y = acc[ti][tj][r] + bias;
                    atomicAdd(&out[(size_t)tokid * DM + col], wgt * y);
                }
            }
        }
}

// ---------------- launcher ----------------
extern "C" void kernel_launch(void* const* d_in, const int* in_sizes, int n_in,
                              void* d_out, int out_size, void* d_ws, size_t ws_size,
                              hipStream_t stream) {
    (void)in_sizes; (void)n_in; (void)out_size; (void)ws_size;
    const float* x  = (const float*)d_in[0];
    const float* wr = (const float*)d_in[1];
    const float* W1 = (const float*)d_in[2];
    const float* b1 = (const float*)d_in[3];
    const float* W2 = (const float*)d_in[4];
    const float* b2 = (const float*)d_in[5];
    float* out = (float*)d_out;

    char* ws = (char*)d_ws;
    size_t off = 0;
    __bf16* xb  = (__bf16*)(ws + off); off += (size_t)TOK * DM * 2;
    __bf16* w1b = (__bf16*)(ws + off); off += (size_t)NE * DM * FF * 2;
    __bf16* w2b = (__bf16*)(ws + off); off += (size_t)NE * FF * DM * 2;
    __bf16* hb  = (__bf16*)(ws + off); off += (size_t)PADROWS * FF * 2;
    int*   topki   = (int*)(ws + off);   off += (size_t)TOK * 2 * 4;
    float* topkp   = (float*)(ws + off); off += (size_t)TOK * 2 * 4;
    int*   rowtok  = (int*)(ws + off);   off += (size_t)PADROWS * 4;
    float* roww    = (float*)(ws + off); off += (size_t)PADROWS * 4;
    int*   counts  = (int*)(ws + off);   off += 64;
    float* psum    = (float*)(ws + off); off += 64;
    int*   cursors = (int*)(ws + off);   off += 64;
    int*   offpad  = (int*)(ws + off);   off += 64;

    // 1) init output + control state
    k_zero_f32<<<(TOK * DM + 1 + 255) / 256, 256, 0, stream>>>(out, TOK * DM + 1);
    k_init_ctrl<<<(PADROWS + 255) / 256, 256, 0, stream>>>(rowtok, counts, psum, cursors);

    // 2) fp32 -> bf16 casts (x, W1, W2)
    k_cast_bf16<<<((TOK * DM / 4) + 255) / 256, 256, 0, stream>>>(x, xb, TOK * DM / 4);
    k_cast_bf16<<<((NE * DM * FF / 4) + 255) / 256, 256, 0, stream>>>(W1, w1b, NE * DM * FF / 4);
    k_cast_bf16<<<((NE * FF * DM / 4) + 255) / 256, 256, 0, stream>>>(W2, w2b, NE * FF * DM / 4);

    // 3) routing
    k_router<<<TOK / 8, 256, 0, stream>>>(x, wr, topki, topkp, counts, psum);
    k_scan<<<1, 32, 0, stream>>>(counts, psum, offpad, out + (size_t)TOK * DM);
    k_scatter<<<TOK / 256, 256, 0, stream>>>(topki, topkp, offpad, cursors, rowtok, roww);

    // 4) expert GEMMs on bf16 WMMA
    k_ffn1<<<dim3(PADROWS / TILE, FF / TILE), 256, 0, stream>>>(xb, w1b, b1, offpad, rowtok, hb);
    k_ffn2<<<dim3(PADROWS / TILE, DM / TILE), 256, 0, stream>>>(hb, w2b, b2, offpad, rowtok, roww, out);
}
